// ConvDecoder_52647709114515
// MI455X (gfx1250) — matrix operand order repaired and checked by reading, not compiled, not run on hardware
//
#include <hip/hip_runtime.h>

// ---------------------------------------------------------------------------
// ConvLSTM decoder on MI455X (gfx1250, wave32) via WMMA bf16 implicit GEMM.
// B=8, T=10, C=64, H=W=64, CH=128.  Compute-bound (~1.27 TFLOP / ~300MB).
// ---------------------------------------------------------------------------

typedef __attribute__((ext_vector_type(16))) __bf16 v16bf;
typedef __attribute__((ext_vector_type(8)))  float  v8f;

union Frag32B { uint4 q[2]; v16bf v; };   // 32-byte A/B fragment (16 bf16)

__device__ __forceinline__ unsigned short f2bf(float f) {
    unsigned int u = __float_as_uint(f);
    unsigned int r = (u + 0x7FFFu + ((u >> 16) & 1u)) >> 16;   // RNE
    return (unsigned short)r;
}
__device__ __forceinline__ v8f vzero8() {
    v8f z = {0.f,0.f,0.f,0.f,0.f,0.f,0.f,0.f};
    return z;
}
__device__ __forceinline__ float sigmoidf_(float x) {
    return 1.0f / (1.0f + __expf(-x));
}

// ---------------------------------------------------------------------------
// Weight repack: fp32 [O][I][3][3]  ->  bf16 [O][9][I]  (K-contiguous for A)
// ---------------------------------------------------------------------------
__global__ void cvt_weights(const float* __restrict__ w,
                            unsigned short* __restrict__ o,
                            int O, int I) {
    int idx = blockIdx.x * 256 + threadIdx.x;
    int total = O * I * 9;
    if (idx >= total) return;
    int i  = idx % I;
    int r  = idx / I;
    int t  = r % 9;
    int oo = r / 9;
    o[idx] = f2bf(w[((size_t)oo * I + i) * 9 + t]);
}

// ---------------------------------------------------------------------------
// Fused ConvLSTM cell:   gates = conv(x,wx)+bx + conv(h,wh)+bh ; LSTM pointwise
//   grid = 1024 blocks (b*128 + y*2 + xhalf), block = 256 (8 waves)
//   wave w: channel tile w (16 hidden ch) for all 4 gates, 2 pixel tiles of 16
//   CX is the x-source channel count (64 for layer0, 128 for layer1).
// ---------------------------------------------------------------------------
template <int CX>
__global__ void __launch_bounds__(256)
lstm_cell_kernel(const float* __restrict__ x, size_t xBatchStride,
                 const float* __restrict__ hprev, const float* __restrict__ cprev,
                 const unsigned short* __restrict__ wx,   // bf16 [512][9][CX]
                 const unsigned short* __restrict__ wh,   // bf16 [512][9][128]
                 const float* __restrict__ bx, const float* __restrict__ bh,
                 float* __restrict__ hout, float* __restrict__ cout) {
    constexpr int HH = 64, WW = 64, CHN = 128;
    constexpr int CXP = CX + 8;          // padded channel stride (mult of 8)
    constexpr int CHP = 136;
    constexpr int hOff = 3 * 36 * CXP;   // hpatch base (ushort index)
    __shared__ __align__(16) unsigned short smem[hOff + 3 * 36 * CHP];

    const int bid = blockIdx.x;
    const int xh  = bid & 1;
    const int y   = (bid >> 1) & 63;
    const int b   = bid >> 7;
    const int x0  = xh * 32;

    const int tid  = threadIdx.x;
    const int wave = tid >> 5;
    const int lane = tid & 31;
    const int m     = lane & 15;   // M row (A) / N col (B,C)
    const int halfk = lane >> 4;

    // ---- stage input + hidden halos into LDS as bf16 [row][col][ch] ----
    {
        const float* sx = x + (size_t)b * xBatchStride;
        int total = 3 * 34 * CX;
        for (int i = tid; i < total; i += 256) {
            int col = i % 34, rem = i / 34;
            int row = rem % 3, c = rem / 3;
            int gy = y + row - 1, gx = x0 + col - 1;
            float v = 0.f;
            if ((unsigned)gy < 64u && (unsigned)gx < 64u)
                v = sx[((size_t)c * HH + gy) * WW + gx];
            smem[(row * 36 + col) * CXP + c] = f2bf(v);
        }
        const float* sh = hprev + (size_t)b * (CHN * HH * WW);
        total = 3 * 34 * CHN;
        for (int i = tid; i < total; i += 256) {
            int col = i % 34, rem = i / 34;
            int row = rem % 3, c = rem / 3;
            int gy = y + row - 1, gx = x0 + col - 1;
            float v = 0.f;
            if ((unsigned)gy < 64u && (unsigned)gx < 64u)
                v = sh[((size_t)c * HH + gy) * WW + gx];
            smem[hOff + (row * 36 + col) * CHP + c] = f2bf(v);
        }
    }
    __syncthreads();

    v8f acc[4][2];
#pragma unroll
    for (int g = 0; g < 4; ++g)
#pragma unroll
        for (int p = 0; p < 2; ++p) acc[g][p] = vzero8();

    // ---- K loop: 9 taps x {x-source, h-source} x 32-wide channel chunks ----
    // Per K-step: batch-load 2 B frags (LDS) + 4 A frags (global, one per
    // gate) into registers first, then issue all 8 WMMAs so load waits
    // retire against descending counters instead of full drains.
    for (int tap = 0; tap < 9; ++tap) {
        const int dx = tap % 3 - 1;
        const int lrow = tap / 3;   // dy+1
#pragma unroll
        for (int src = 0; src < 2; ++src) {
            const unsigned short* wsrc = src ? wh : wx;
            const int Ci = src ? CHN : CX;
            const int CP = src ? CHP : CXP;
            const int po = src ? hOff : 0;
#pragma unroll
            for (int kc = 0; kc < (src ? CHN : CX); kc += 32) {
                Frag32B bfr[2];
#pragma unroll
                for (int pt = 0; pt < 2; ++pt) {
                    int col = pt * 16 + m + dx + 1;
                    int base = po + (lrow * 36 + col) * CP + kc + 16 * halfk;
                    bfr[pt].q[0] = *(const uint4*)(smem + base);
                    bfr[pt].q[1] = *(const uint4*)(smem + base + 8);
                }
                Frag32B afr[4];
#pragma unroll
                for (int g = 0; g < 4; ++g) {
                    int oc = g * 128 + wave * 16 + m;
                    const unsigned short* wp =
                        wsrc + ((size_t)oc * 9 + tap) * Ci + kc + 16 * halfk;
                    afr[g].q[0] = *(const uint4*)(wp);
                    afr[g].q[1] = *(const uint4*)(wp + 8);
                }
#pragma unroll
                for (int g = 0; g < 4; ++g)
#pragma unroll
                    for (int pt = 0; pt < 2; ++pt)
                        acc[g][pt] = __builtin_amdgcn_wmma_f32_16x16x32_bf16(
                            false, afr[g].v, false, bfr[pt].v,
                            (short)0, acc[g][pt], false, false);
            }
        }
    }

    // ---- fused LSTM pointwise epilogue ----
    float bias[4][8];
#pragma unroll
    for (int g = 0; g < 4; ++g)
#pragma unroll
        for (int r = 0; r < 8; ++r) {
            int ch = wave * 16 + r + 8 * halfk;
            int oc = g * 128 + ch;
            bias[g][r] = bx[oc] + bh[oc];
        }
#pragma unroll
    for (int pt = 0; pt < 2; ++pt) {
        int px = x0 + pt * 16 + m;
#pragma unroll
        for (int r = 0; r < 8; ++r) {
            int ch = wave * 16 + r + 8 * halfk;
            size_t idx = (((size_t)b * CHN + ch) * HH + y) * WW + px;
            float iv = sigmoidf_(acc[0][pt][r] + bias[0][r]);
            float fv = sigmoidf_(acc[1][pt][r] + bias[1][r]);
            float gv = tanhf    (acc[2][pt][r] + bias[2][r]);
            float ov = sigmoidf_(acc[3][pt][r] + bias[3][r]);
            float cy = fv * cprev[idx] + iv * gv;
            cout[idx] = cy;
            hout[idx] = ov * tanhf(cy);
        }
    }
}

// ---------------------------------------------------------------------------
// Top conv: out[:,t] = conv(h, wtop) + btop.  O=64, I=128.
//   grid = 1024, block = 128 (4 waves); wave = oc tile, 32 pixels / block
// ---------------------------------------------------------------------------
__global__ void __launch_bounds__(128)
top_conv_kernel(const float* __restrict__ h,
                const unsigned short* __restrict__ wt,   // bf16 [64][9][128]
                const float* __restrict__ bt,
                float* __restrict__ out, int tstep) {
    constexpr int HH = 64, WW = 64, CHN = 128, CHP = 136, TT = 10, CO = 64;
    __shared__ __align__(16) unsigned short smem[3 * 36 * 136];

    const int bid = blockIdx.x;
    const int xh  = bid & 1;
    const int y   = (bid >> 1) & 63;
    const int b   = bid >> 7;
    const int x0  = xh * 32;

    const int tid  = threadIdx.x;
    const int wave = tid >> 5;
    const int lane = tid & 31;
    const int m     = lane & 15;
    const int halfk = lane >> 4;

    {
        const float* sh = h + (size_t)b * (CHN * HH * WW);
        int total = 3 * 34 * CHN;
        for (int i = tid; i < total; i += 128) {
            int col = i % 34, rem = i / 34;
            int row = rem % 3, c = rem / 3;
            int gy = y + row - 1, gx = x0 + col - 1;
            float v = 0.f;
            if ((unsigned)gy < 64u && (unsigned)gx < 64u)
                v = sh[((size_t)c * HH + gy) * WW + gx];
            smem[(row * 36 + col) * CHP + c] = f2bf(v);
        }
    }
    __syncthreads();

    v8f acc[2] = {vzero8(), vzero8()};

    for (int tap = 0; tap < 9; ++tap) {
        const int dx = tap % 3 - 1;
        const int lrow = tap / 3;
#pragma unroll
        for (int kc = 0; kc < CHN; kc += 32) {
            Frag32B bfr[2];
#pragma unroll
            for (int pt = 0; pt < 2; ++pt) {
                int col = pt * 16 + m + dx + 1;
                int base = (lrow * 36 + col) * CHP + kc + 16 * halfk;
                bfr[pt].q[0] = *(const uint4*)(smem + base);
                bfr[pt].q[1] = *(const uint4*)(smem + base + 8);
            }
            int oc = wave * 16 + m;
            const unsigned short* wp =
                wt + ((size_t)oc * 9 + tap) * CHN + kc + 16 * halfk;
            Frag32B afr;
            afr.q[0] = *(const uint4*)(wp);
            afr.q[1] = *(const uint4*)(wp + 8);
#pragma unroll
            for (int pt = 0; pt < 2; ++pt)
                acc[pt] = __builtin_amdgcn_wmma_f32_16x16x32_bf16(
                    false, afr.v, false, bfr[pt].v,
                    (short)0, acc[pt], false, false);
        }
    }

#pragma unroll
    for (int pt = 0; pt < 2; ++pt) {
        int px = x0 + pt * 16 + m;
#pragma unroll
        for (int r = 0; r < 8; ++r) {
            int ch = wave * 16 + r + 8 * halfk;
            size_t idx = (((size_t)b * TT + tstep) * CO + ch) * (HH * WW)
                       + (size_t)y * WW + px;
            out[idx] = acc[pt][r] + bt[ch];
        }
    }
}

// ---------------------------------------------------------------------------
extern "C" void kernel_launch(void* const* d_in, const int* in_sizes, int n_in,
                              void* d_out, int out_size, void* d_ws, size_t ws_size,
                              hipStream_t stream) {
    (void)in_sizes; (void)n_in; (void)out_size; (void)ws_size;
    const float* target = (const float*)d_in[0];
    const float* h0i = (const float*)d_in[1];
    const float* c0i = (const float*)d_in[2];
    const float* h1i = (const float*)d_in[3];
    const float* c1i = (const float*)d_in[4];
    const float* wi0 = (const float*)d_in[5];
    const float* bi0 = (const float*)d_in[6];
    const float* wh0 = (const float*)d_in[7];
    const float* bh0 = (const float*)d_in[8];
    const float* wi1 = (const float*)d_in[9];
    const float* bi1 = (const float*)d_in[10];
    const float* wh1 = (const float*)d_in[11];
    const float* bh1 = (const float*)d_in[12];
    const float* wtp = (const float*)d_in[13];
    const float* btp = (const float*)d_in[14];
    float* out = (float*)d_out;

    char* ws = (char*)d_ws;
    // bf16 weight buffers
    unsigned short* wbi0 = (unsigned short*)(ws);                  // 512*64*9
    unsigned short* wbh0 = (unsigned short*)(ws + 589824);         // 512*128*9
    unsigned short* wbi1 = (unsigned short*)(ws + 1769472);
    unsigned short* wbh1 = (unsigned short*)(ws + 2949120);
    unsigned short* wbtp = (unsigned short*)(ws + 4128768);        // 64*128*9
    // fp32 ping-pong state buffers, each 8*128*64*64*4 = 16 MiB
    const size_t SB = (size_t)8 * 128 * 64 * 64 * 4;
    char* sbase = ws + 4276224;
    float* h0b[2] = {(float*)(sbase + 0 * SB), (float*)(sbase + 1 * SB)};
    float* c0b[2] = {(float*)(sbase + 2 * SB), (float*)(sbase + 3 * SB)};
    float* h1b[2] = {(float*)(sbase + 4 * SB), (float*)(sbase + 5 * SB)};
    float* c1b[2] = {(float*)(sbase + 6 * SB), (float*)(sbase + 7 * SB)};

    // repack weights fp32 [O][I][3][3] -> bf16 [O][9][I]
    cvt_weights<<<(512 * 64 * 9 + 255) / 256, 256, 0, stream>>>(wi0, wbi0, 512, 64);
    cvt_weights<<<(512 * 128 * 9 + 255) / 256, 256, 0, stream>>>(wh0, wbh0, 512, 128);
    cvt_weights<<<(512 * 128 * 9 + 255) / 256, 256, 0, stream>>>(wi1, wbi1, 512, 128);
    cvt_weights<<<(512 * 128 * 9 + 255) / 256, 256, 0, stream>>>(wh1, wbh1, 512, 128);
    cvt_weights<<<(64 * 128 * 9 + 255) / 256, 256, 0, stream>>>(wtp, wbtp, 64, 128);

    const size_t HW = 64 * 64;

    // out[:,0] = conv(h1_init, wtop)
    top_conv_kernel<<<1024, 128, 0, stream>>>(h1i, wbtp, btp, out, 0);

    for (int s = 0; s < 9; ++s) {
        const float* xsrc = target + (size_t)s * 64 * HW;   // [B,T,64,H,W] slice t=s
        const size_t xbs = (size_t)10 * 64 * HW;
        const float* h0in = s ? (const float*)h0b[(s - 1) & 1] : h0i;
        const float* c0in = s ? (const float*)c0b[(s - 1) & 1] : c0i;
        const float* h1in = s ? (const float*)h1b[(s - 1) & 1] : h1i;
        const float* c1in = s ? (const float*)c1b[(s - 1) & 1] : c1i;

        // layer 0: x1 = h0b[s&1], c0 -> c0b[s&1]
        lstm_cell_kernel<64><<<1024, 256, 0, stream>>>(
            xsrc, xbs, h0in, c0in, wbi0, wbh0, bi0, bh0,
            h0b[s & 1], c0b[s & 1]);
        // layer 1: x2 = h1b[s&1]
        lstm_cell_kernel<128><<<1024, 256, 0, stream>>>(
            h0b[s & 1], (size_t)128 * HW, h1in, c1in, wbi1, wbh1, bi1, bh1,
            h1b[s & 1], c1b[s & 1]);
        // out[:, s+1] = conv(x2, wtop)
        top_conv_kernel<<<1024, 128, 0, stream>>>(h1b[s & 1], wbtp, btp, out, s + 1);
    }
}
